// FLTAttention_8976481649324
// MI455X (gfx1250) — matrix operand order repaired and checked by reading, not compiled, not run on hardware
//
#include <hip/hip_runtime.h>
#include <math.h>

// ---------------- types / constants ----------------
typedef _Float16 f16;
typedef __attribute__((ext_vector_type(16))) _Float16 v16h;
typedef __attribute__((ext_vector_type(8)))  _Float16 v8h;
typedef __attribute__((ext_vector_type(8)))  float    v8f;
typedef __attribute__((ext_vector_type(4)))  unsigned int u32x4;
typedef __attribute__((ext_vector_type(8)))  int i32x8;
typedef __attribute__((ext_vector_type(4)))  int i32x4;

#define HH    8
#define DHD   64
#define NN    8192L
#define BBAT  4L
#define MFEAT 256
#define FDIM  192
#define RATIO   0.0625f                 // 256^-0.5
#define SM_EPS  1e-6f
#define NRM_EPS 1e-6f
#define QSCL    0.3535533905932738f     // sqrt(1/8)
#define RFFSCL  0.1767766952966369f     // sqrt(2/64)

// TDM path available? (device pass; host pass falls back)
#if defined(__has_builtin)
#if __has_builtin(__builtin_amdgcn_tensor_load_to_lds) && __has_builtin(__builtin_amdgcn_s_wait_tensorcnt)
#define USE_TDM 1
#else
#define USE_TDM 0
#endif
#else
#define USE_TDM 0
#endif

union HU { v16h v; v8h h[2]; f16 e[16]; };

__device__ __forceinline__ v8f zero8() {
  v8f z = {0.f,0.f,0.f,0.f,0.f,0.f,0.f,0.f};
  return z;
}

__device__ __forceinline__ v8f wmma16(v16h a, v16h b, v8f c) {
  // D = A(16x32 f16) * B(32x16 f16) + C(16x16 f32)
  return __builtin_amdgcn_wmma_f32_16x16x32_f16(false, a, false, b, (short)0, c, false, false);
}

// A fragment: rowptr points at the start of row (m0 + lane%16); K-contiguous storage.
// halves 0..7  <- K = k0 + (lane/16)*16 + 0..7
// halves 8..15 <- K = k0 + (lane/16)*16 + 8 + 0..7
__device__ __forceinline__ v16h load_a_frag(const f16* rowptr, int k0, int lane) {
  const f16* p = rowptr + k0 + ((lane >> 4) << 4);
  HU u;
  u.h[0] = *(const v8h*)(p);
  u.h[1] = *(const v8h*)(p + 8);
  return u.v;
}

#if USE_TDM
// 2-D TDM: DMA (dim1 rows) x (dim0 elems) of 2-byte data, row stride stride0 elems,
// from global into LDS at byte offset lds_off. D# layout per CDNA5 ISA ch.8.
// This toolchain's builtin takes 6 args (g0, g1, g2, g3, extra group, cpol).
__device__ __forceinline__ void tdm_load_2d_to_lds(const void* gsrc, unsigned lds_off,
                                                   unsigned dim0, unsigned dim1,
                                                   unsigned stride0) {
  unsigned long long ga = (unsigned long long)(size_t)gsrc;
  u32x4 g0;
  g0[0] = 1u;                                             // count=1, user descriptor
  g0[1] = lds_off;                                        // lds_addr (bytes)
  g0[2] = (unsigned)(ga & 0xFFFFFFFFu);                   // global_addr[31:0]
  g0[3] = (unsigned)((ga >> 32) & 0x01FFFFFFu)            // global_addr[56:32]
        | (2u << 30);                                     // type=2 ("image")
  i32x8 g1;
  g1[0] = (int)(1u << 16);                                // data_size=1 -> 2 bytes
  g1[1] = (int)((dim0 & 0xFFFFu) << 16);                  // tensor_dim0[15:0] @ bits 63:48
  g1[2] = (int)(((dim0 >> 16) & 0xFFFFu)                  // tensor_dim0[31:16]
        | ((dim1 & 0xFFFFu) << 16));                      // tensor_dim1[15:0]
  g1[3] = (int)(((dim1 >> 16) & 0xFFFFu)                  // tensor_dim1[31:16]
        | ((dim0 & 0xFFFFu) << 16));                      // tile_dim0 = dim0
  g1[4] = (int)(dim1 & 0xFFFFu);                          // tile_dim1 = dim1; tile_dim2=0
  g1[5] = (int)stride0;                                   // tensor_dim0_stride[31:0]
  g1[6] = 0;                                              // stride0 hi / dim1_stride lo
  g1[7] = 0;
  i32x4 z4 = {0, 0, 0, 0};
  i32x8 z8 = {0, 0, 0, 0, 0, 0, 0, 0};
  __builtin_amdgcn_tensor_load_to_lds(g0, g1, z4, z4, z8, 0);
}
#endif

// order-preserving float<->uint for atomicMax on signed floats
__device__ __forceinline__ unsigned enc_max(float f) {
  unsigned u = __float_as_uint(f);
  return (u & 0x80000000u) ? ~u : (u | 0x80000000u);
}
__device__ __forceinline__ float dec_max(unsigned u) {
  return __uint_as_float((u & 0x80000000u) ? (u ^ 0x80000000u) : ~u);
}
__device__ __forceinline__ float wave_max(float x) {
  for (int m = 16; m; m >>= 1) x = fmaxf(x, __shfl_xor(x, m, 32));
  return x;
}
__device__ __forceinline__ float wave_sum(float x) {
  for (int m = 16; m; m >>= 1) x += __shfl_xor(x, m, 32);
  return x;
}

// ---------------- K0: per-head constants + mkey init ----------------
// cst[0..15]  = sqrt(alpha[h][r])       (h*2 + r)
// cst[16..39] = sqrt(new_qw[h][0..2])   (16 + h*3 + i)
__global__ void k_head_consts(const float* __restrict__ w_rpe, float* cst, unsigned* mkey) {
  __shared__ float es[128];
  int t = threadIdx.x;                 // 128 threads: h*16 + (r*8 + c*4 + k)
  int h = t >> 4, rem = t & 15;
  float s = 0.f;
  for (int d = 0; d < DHD; ++d) s += w_rpe[(h*DHD + d)*16 + rem];
  es[t] = __expf(fminf(s, 50.f));
  __syncthreads();
  if ((t & 3) == 0) {
    float e = es[t] + es[t+1] + es[t+2] + es[t+3];
    int r = (rem >> 3) & 1, c = (rem >> 2) & 1;
    float se = sqrtf(e);
    if (c == 0) {
      cst[h*2 + r] = se;                        // sqrt(alpha)
    } else {
      if (r == 0) { cst[16 + h*3 + 0] = se; cst[16 + h*3 + 1] = se; }
      else        { cst[16 + h*3 + 2] = se; }   // sqrt(new_qw)
    }
  }
  if (t < 32) mkey[t] = 0u;            // identity for encoded atomicMax
}

// ---------------- f32 -> f16 convert ----------------
__global__ void k_f32_to_f16(const float* __restrict__ src, f16* __restrict__ dst, int n) {
  int i = blockIdx.x * blockDim.x + threadIdx.x;
  if (i < n) dst[i] = (f16)src[i];
}

// ---------------- K1: featurize (phi via RFF, f16 staging, diags, vT) ----------------
__global__ void k_featurize(const float* __restrict__ query, const float* __restrict__ key,
                            const float* __restrict__ value, const float* __restrict__ coords,
                            const float* __restrict__ omega_dR, const float* __restrict__ omega_dA,
                            const float* __restrict__ cst,
                            f16* __restrict__ qh, f16* __restrict__ kh,
                            f16* __restrict__ phih, f16* __restrict__ vT,
                            float* __restrict__ diag_q, float* __restrict__ diag_k) {
  long tid = (long)blockIdx.x * blockDim.x + threadIdx.x;
  int lane = (int)(tid & 31);
  long row = tid >> 5;                 // ((b*8 + h)*8192 + n)
  long n  = row & (NN - 1);
  long bh = row >> 13;
  int  h  = (int)(bh & 7);
  long bb = bh >> 3;

  float sA0 = cst[h*2 + 0], sA1 = cst[h*2 + 1];
  float sw0 = cst[16 + h*3 + 0], sw1 = cst[16 + h*3 + 1], sw2 = cst[16 + h*3 + 2];

  const float* cp = coords + (bb*NN + n)*3;
  float c0 = cp[0]*sw0, c1 = cp[1]*sw1, c2 = cp[2]*sw2;

  float u0 = c0*omega_dR[lane] + c1*omega_dR[32 + lane];
  float u1 = c2*omega_dA[lane];
  float p0c = cosf(u0)*RFFSCL*sA0, p0s = sinf(u0)*RFFSCL*sA0;
  float p1c = cosf(u1)*RFFSCL*sA1, p1s = sinf(u1)*RFFSCL*sA1;

  const float* qp = query + (bb*NN + n)*512 + h*DHD;
  const float* kp = key   + (bb*NN + n)*512 + h*DHD;
  const float* vp = value + (bb*NN + n)*512 + h*DHD;
  float q0 = qp[lane]*QSCL,  q1 = qp[lane+32]*QSCL;
  float k0 = kp[lane]*QSCL,  k1 = kp[lane+32]*QSCL;

  f16* qr = qh + row*64;  qr[lane] = (f16)q0;  qr[lane+32] = (f16)q1;
  f16* kr = kh + row*64;  kr[lane] = (f16)k0;  kr[lane+32] = (f16)k1;
  f16* pr = phih + row*128;
  pr[lane]    = (f16)p0c;  pr[lane+32] = (f16)p0s;
  pr[lane+64] = (f16)p1c;  pr[lane+96] = (f16)p1s;

  // vT[bh][e][n] with ones column e=64, zero padding e=65..79 (k_sum trick)
  f16* vtb = vT + bh*80*NN;
  vtb[(long)lane*NN + n]        = (f16)vp[lane];
  vtb[(long)(lane+32)*NN + n]   = (f16)vp[lane+32];
  if (lane < 16) vtb[(long)(64+lane)*NN + n] = (f16)((lane == 0) ? 1.0f : 0.0f);

  float phsq = p0c*p0c + p0s*p0s + p1c*p1c + p1s*p1s;
  float sq = wave_sum(q0*q0 + q1*q1 + phsq);
  float sk = wave_sum(k0*k0 + k1*k1 + phsq);
  if (lane == 0) { diag_q[row] = 0.5f*sq; diag_k[row] = 0.5f*sk; }
}

// ---------------- K2a: key dash pass 1 — global max per (b,h) ----------------
// proj^T (96KB) staged to LDS once per workgroup via Tensor Data Mover.
__global__ void k_key_max(const f16* __restrict__ kh, const f16* __restrict__ phih,
                          const f16* __restrict__ projh, unsigned* __restrict__ mkey) {
  extern __shared__ __align__(16) char smem_raw[];
  f16* sproj = (f16*)smem_raw;
  int lane = threadIdx.x & 31;
  long wave = (long)blockIdx.x * (blockDim.x >> 5) + (threadIdx.x >> 5);
  long row0 = wave * 16;
  long bh = row0 >> 13;
  long ra = row0 + (lane & 15);

#if USE_TDM
  if (threadIdx.x < 32) {            // wave 0 issues the DMA (EXEC ignored by TDM)
    tdm_load_2d_to_lds(projh, (unsigned)(size_t)sproj, FDIM, MFEAT, FDIM);
    __builtin_amdgcn_s_wait_tensorcnt(0);
  }
  __syncthreads();
  const f16* pbase = sproj;
#else
  const f16* pbase = projh;
#endif

  v16h afr[6];
  const f16* kr = kh + ra*64;
  afr[0] = load_a_frag(kr, 0, lane);
  afr[1] = load_a_frag(kr, 32, lane);
  const f16* pr = phih + ra*128;
  for (int t = 0; t < 4; ++t) afr[2+t] = load_a_frag(pr, t*32, lane);

  float mx = -3.0e38f;
  for (int nt = 0; nt < 16; ++nt) {
    v8f acc = zero8();
    const f16* bcol = pbase + (long)(nt*16 + (lane & 15))*FDIM + ((lane >> 4) << 4);
    for (int kt = 0; kt < 6; ++kt)
      acc = wmma16(afr[kt], *(const v16h*)(bcol + kt*32), acc);
    for (int i = 0; i < 8; ++i) mx = fmaxf(mx, acc[i]);
  }
  mx = wave_max(mx);
  if (lane == 0) atomicMax(&mkey[bh], enc_max(mx));
}

// ---------------- K2b: key dash pass 2 — kfeatT[bh][f][n] (f16) ----------------
__global__ void k_key_feat(const f16* __restrict__ kh, const f16* __restrict__ phih,
                           const f16* __restrict__ projh, const float* __restrict__ diag_k,
                           const unsigned char* __restrict__ mask,
                           const unsigned* __restrict__ mkey, f16* __restrict__ kfeatT) {
  extern __shared__ __align__(16) char smem_raw[];
  f16* sproj = (f16*)smem_raw;
  int lane = threadIdx.x & 31;
  long wave = (long)blockIdx.x * (blockDim.x >> 5) + (threadIdx.x >> 5);
  long row0 = wave * 16;
  long bh = row0 >> 13;
  long ra = row0 + (lane & 15);

#if USE_TDM
  if (threadIdx.x < 32) {
    tdm_load_2d_to_lds(projh, (unsigned)(size_t)sproj, FDIM, MFEAT, FDIM);
    __builtin_amdgcn_s_wait_tensorcnt(0);
  }
  __syncthreads();
  const f16* pbase = sproj;
#else
  const f16* pbase = projh;
#endif

  v16h afr[6];
  const f16* kr = kh + ra*64;
  afr[0] = load_a_frag(kr, 0, lane);
  afr[1] = load_a_frag(kr, 32, lane);
  const f16* pr = phih + ra*128;
  for (int t = 0; t < 4; ++t) afr[2+t] = load_a_frag(pr, t*32, lane);

  float m = dec_max(mkey[bh]);
  long rbase = row0 + ((lane >> 4) << 3);      // 8 consecutive rows this lane covers
  long nb = rbase & (NN - 1);
  long bb = rbase >> 16;                       // rows per batch = H*N = 65536
  float dk[8], mf[8];
  for (int rr = 0; rr < 8; ++rr) {
    dk[rr] = diag_k[rbase + rr];
    mf[rr] = mask[bb*NN + nb + rr] ? 1.0f : 0.0f;
  }

  f16* ktb = kfeatT + bh*(long)MFEAT*NN;
  long n0 = row0 & (NN - 1);
  for (int nt = 0; nt < 16; ++nt) {
    v8f acc = zero8();
    const f16* bcol = pbase + (long)(nt*16 + (lane & 15))*FDIM + ((lane >> 4) << 4);
    for (int kt = 0; kt < 6; ++kt)
      acc = wmma16(afr[kt], *(const v16h*)(bcol + kt*32), acc);
    int j = nt*16 + (lane & 15);
    v8h o;
    for (int rr = 0; rr < 8; ++rr) {
      float kf = RATIO * (__expf(acc[rr] - dk[rr] - m) + SM_EPS) * mf[rr];
      o[rr] = (f16)kf;
    }
    *(v8h*)(ktb + (long)j*NN + n0 + ((lane >> 4) << 3)) = o;   // contiguous in n
  }
}

// ---------------- K3: ctx GEMM — ctxT[bh][e(80)][f(256)] = (kfeatT * vT)^T ----------------
__global__ void k_ctx(const f16* __restrict__ kfeatT, const f16* __restrict__ vT,
                      f16* __restrict__ ctxT) {
  int lane = threadIdx.x & 31;
  long wave = (long)blockIdx.x * (blockDim.x >> 5) + (threadIdx.x >> 5);
  long bh = wave >> 4;
  int  mt = (int)(wave & 15);
  const f16* A  = kfeatT + bh*(long)MFEAT*NN + (long)(mt*16 + (lane & 15))*NN;
  const f16* Bb = vT + bh*80*NN;

  v8f acc[5];
  for (int et = 0; et < 5; ++et) acc[et] = zero8();

  for (int kt = 0; kt < 256; ++kt) {
    int k0 = kt*32;
    __builtin_prefetch(A + k0 + 512, 0, 1);       // global_prefetch_b8
    v16h a = load_a_frag(A, k0, lane);
    for (int et = 0; et < 5; ++et) {
      v16h b = *(const v16h*)(Bb + (long)(et*16 + (lane & 15))*NN + k0 + ((lane >> 4) << 4));
      acc[et] = wmma16(a, b, acc[et]);
    }
  }
  for (int et = 0; et < 5; ++et) {
    int e = et*16 + (lane & 15);
    v8h o;
    for (int rr = 0; rr < 8; ++rr) o[rr] = (f16)acc[et][rr];
    *(v8h*)(ctxT + (bh*80 + e)*(long)MFEAT + mt*16 + ((lane >> 4) << 3)) = o;
  }
}

// ---------------- K4: fused query path — dash -> rowmax -> qfeat -> out = ctx*q*dinv ----------------
__global__ void __launch_bounds__(64)
k_query_out(const f16* __restrict__ qh, const f16* __restrict__ phih,
            const f16* __restrict__ projh, const f16* __restrict__ ctxT,
            const float* __restrict__ diag_q, f16* __restrict__ outh) {
  __shared__ float dash[2][16][256];
  __shared__ float dinv[2][16];
  int lane = threadIdx.x & 31;
  int w    = threadIdx.x >> 5;
  long wave = (long)blockIdx.x * 2 + w;
  long row0 = wave * 16;
  long bh = row0 >> 13;
  int  h  = (int)(bh & 7);
  long bb = bh >> 3;
  long ra = row0 + (lane & 15);

  v16h afr[6];
  const f16* qr = qh + ra*64;
  afr[0] = load_a_frag(qr, 0, lane);
  afr[1] = load_a_frag(qr, 32, lane);
  const f16* pr = phih + ra*128;
  for (int t = 0; t < 4; ++t) afr[2+t] = load_a_frag(pr, t*32, lane);

  // dash = qaug @ proj^T  (M=16 rows, N=256, K=192) -> LDS
  for (int nt = 0; nt < 16; ++nt) {
    v8f acc = zero8();
    const f16* bcol = projh + (long)(nt*16 + (lane & 15))*FDIM + ((lane >> 4) << 4);
    for (int kt = 0; kt < 6; ++kt)
      acc = wmma16(afr[kt], *(const v16h*)(bcol + kt*32), acc);
    int col = nt*16 + (lane & 15);
    int mr  = (lane >> 4) * 8;
    for (int rr = 0; rr < 8; ++rr) dash[w][mr + rr][col] = acc[rr];
  }
  __syncthreads();

  // rowwise max (query mode), then qfeat A fragments (f16)
  int myrow = lane & 15;
  float mrow = -3.0e38f;
  for (int j = 0; j < 256; ++j) mrow = fmaxf(mrow, dash[w][myrow][j]);
  float dq = diag_q[row0 + myrow];

  v16h qf[8];
  for (int kt2 = 0; kt2 < 8; ++kt2) {
    HU u;
    for (int p = 0; p < 16; ++p) {
      int f = kt2*32 + ((lane >> 4) << 4) + p;
      float val = RATIO * (__expf(dash[w][myrow][f] - dq - mrow) + SM_EPS);
      u.e[p] = (f16)val;
    }
    qf[kt2] = u.v;
  }

  const f16* ctxb = ctxT + bh*80*(long)MFEAT;

  // e-tile [64..79] first: column e=64 is q . k_sum -> denominator
  {
    v8f accd = zero8();
    const f16* bcol = ctxb + (long)(64 + (lane & 15))*MFEAT + ((lane >> 4) << 4);
    for (int kt2 = 0; kt2 < 8; ++kt2)
      accd = wmma16(qf[kt2], *(const v16h*)(bcol + kt2*32), accd);
    if ((lane & 15) == 0) {
      int mr = (lane >> 4) * 8;
      for (int rr = 0; rr < 8; ++rr) dinv[w][mr + rr] = 1.0f / (accd[rr] + NRM_EPS);
    }
  }
  __syncthreads();

  long n0 = row0 & (NN - 1);
  for (int et = 0; et < 4; ++et) {
    v8f acc = zero8();
    const f16* bcol = ctxb + (long)(et*16 + (lane & 15))*MFEAT + ((lane >> 4) << 4);
    for (int kt2 = 0; kt2 < 8; ++kt2)
      acc = wmma16(qf[kt2], *(const v16h*)(bcol + kt2*32), acc);
    int e = et*16 + (lane & 15);
    for (int rr = 0; rr < 8; ++rr) {
      int M = (lane >> 4)*8 + rr;
      float o = acc[rr] * dinv[w][M];
      outh[(bb*NN + n0 + M)*512 + h*DHD + e] = (f16)o;
    }
  }
}

// ---------------- K5: final projection  out = outh @ out_w^T + out_b ----------------
__global__ void k_out_proj(const f16* __restrict__ outh, const f16* __restrict__ outwh,
                           const float* __restrict__ out_b, float* __restrict__ out) {
  int lane = threadIdx.x & 31;
  long wave = (long)blockIdx.x * (blockDim.x >> 5) + (threadIdx.x >> 5);
  long m0 = wave * 16;
  const f16* Arow = outh + (m0 + (lane & 15))*512;
  for (int et = 0; et < 4; ++et) {
    v8f acc = zero8();
    int e = et*16 + (lane & 15);
    const f16* Bc = outwh + (long)e*512 + ((lane >> 4) << 4);
    for (int kt = 0; kt < 16; ++kt) {
      v16h a = load_a_frag(Arow, kt*32, lane);
      acc = wmma16(a, *(const v16h*)(Bc + kt*32), acc);
    }
    float bias = out_b[e];
    for (int rr = 0; rr < 8; ++rr) {
      long M = m0 + (lane >> 4)*8 + rr;
      out[M*64 + e] = acc[rr] + bias;
    }
  }
}

// ---------------- host launcher ----------------
extern "C" void kernel_launch(void* const* d_in, const int* in_sizes, int n_in,
                              void* d_out, int out_size, void* d_ws, size_t ws_size,
                              hipStream_t stream) {
  const float* query  = (const float*)d_in[0];
  const float* key    = (const float*)d_in[1];
  const float* value  = (const float*)d_in[2];
  const float* coords = (const float*)d_in[3];
  const unsigned char* mask = (const unsigned char*)d_in[4];  // bool array
  const float* w_rpe  = (const float*)d_in[5];
  const float* omega_dR = (const float*)d_in[6];
  const float* omega_dA = (const float*)d_in[7];
  const float* proj   = (const float*)d_in[8];
  const float* out_w  = (const float*)d_in[9];
  const float* out_b  = (const float*)d_in[10];
  float* out = (float*)d_out;
  (void)in_sizes; (void)n_in; (void)out_size; (void)ws_size;

  char* ws = (char*)d_ws;
  size_t off = 0;
  auto alloc = [&](size_t bytes) -> char* {
    char* p = ws + off;
    off = (off + bytes + 255) & ~(size_t)255;
    return p;
  };
  const long ROWS = BBAT * HH * NN;                     // 262144
  float*    cst    = (float*)   alloc(256);
  unsigned* mkey   = (unsigned*)alloc(256);
  f16*      projh  = (f16*)alloc((size_t)MFEAT*FDIM*2); // 96 KB
  f16*      outwh  = (f16*)alloc((size_t)64*512*2);
  f16*      qh     = (f16*)alloc((size_t)ROWS*64*2);
  f16*      kh     = (f16*)alloc((size_t)ROWS*64*2);
  f16*      phih   = (f16*)alloc((size_t)ROWS*128*2);
  f16*      vT     = (f16*)alloc((size_t)32*80*NN*2);
  f16*      kfeatT = (f16*)alloc((size_t)32*MFEAT*NN*2);
  f16*      ctxT   = (f16*)alloc((size_t)32*80*MFEAT*2);
  float*    diag_q = (float*)alloc((size_t)ROWS*4);
  float*    diag_k = (float*)alloc((size_t)ROWS*4);
  f16*      outh   = (f16*)alloc((size_t)BBAT*NN*512*2);

  const size_t PROJ_LDS = (size_t)MFEAT * FDIM * sizeof(f16);  // 96 KB dynamic LDS

  k_head_consts<<<1, 128, 0, stream>>>(w_rpe, cst, mkey);
  k_f32_to_f16<<<(MFEAT*FDIM + 255)/256, 256, 0, stream>>>(proj, projh, MFEAT*FDIM);
  k_f32_to_f16<<<(64*512 + 255)/256, 256, 0, stream>>>(out_w, outwh, 64*512);

  k_featurize<<<(unsigned)(ROWS*32/256), 256, 0, stream>>>(
      query, key, value, coords, omega_dR, omega_dA, cst,
      qh, kh, phih, vT, diag_q, diag_k);

  // 16384 waves total (16 rows each), 4 waves per block; projh DMA'd to LDS per block
  k_key_max <<<4096, 128, PROJ_LDS, stream>>>(kh, phih, projh, mkey);
  k_key_feat<<<4096, 128, PROJ_LDS, stream>>>(kh, phih, projh, diag_k, mask, mkey, kfeatT);

  // 512 waves (32 bh * 16 mtiles), 4 per block
  k_ctx<<<128, 128, 0, stream>>>(kfeatT, vT, ctxT);

  // 16384 waves, 2 per block (16KB LDS dash tile per wave)
  k_query_out<<<8192, 64, 0, stream>>>(qh, phih, projh, ctxT, diag_q, outh);

  // 2048 waves, 4 per block
  k_out_proj<<<512, 128, 0, stream>>>(outh, outwh, out_b, out);
}
